// CrossAttention_55473797595555
// MI455X (gfx1250) — matrix operand order repaired and checked by reading, not compiled, not run on hardware
//
#include <hip/hip_runtime.h>
#include <hip/hip_bf16.h>

// ---------------------------------------------------------------------------
// CrossAttention for MI455X (gfx1250, wave32, WMMA 16x16x32 f16 -> f32)
//   Q = x Wq^T + bq ; K = c Wk^T + bk ; V = c Wv^T + bv   (f16 to d_ws)
//   out = softmax(Q K^T / sqrt(D)) V        (flash, async-LDS K feed, f32 out)
// ---------------------------------------------------------------------------

typedef _Float16 half_t;
typedef __attribute__((ext_vector_type(8)))  _Float16 v8h;
typedef __attribute__((ext_vector_type(16))) _Float16 v16h;
typedef __attribute__((ext_vector_type(8)))  float    v8f;

static constexpr int Dm  = 512;   // model dim
static constexpr int NQ  = 4096;
static constexpr int NK  = 4096;
static constexpr int Bsz = 8;

// ---- CDNA5 async global->LDS DMA via inline asm (ISA 15.18.3, op 98) ------
// vDST = per-lane LDS byte offset (hardware adds the wave's LDS_BASE),
// vADDR = 64-bit global byte address. Tracked by ASYNCcnt.
__device__ __forceinline__ void async_copy16(const half_t* g, unsigned lds_off) {
  asm volatile("global_load_async_to_lds_b128 %0, %1, off"
               :: "v"(lds_off), "v"((unsigned long long)(uintptr_t)g)
               : "memory");
}
#define WAIT_ASYNC(n) asm volatile("s_wait_asynccnt %0" ::"n"(n) : "memory")

__device__ __forceinline__ v8f zero8() {
  v8f z;
#pragma unroll
  for (int i = 0; i < 8; ++i) z[i] = 0.0f;
  return z;
}

__device__ __forceinline__ v16h combine16(v8h a, v8h b) {
  return __builtin_shufflevector(a, b, 0,1,2,3,4,5,6,7,8,9,10,11,12,13,14,15);
}

// A-matrix fragment (16x32 f16). p = row base + chunk base + (lane>>4)*16.
// halves 0..15 <-> k = kh*16 + h (contiguous 32 bytes)
__device__ __forceinline__ v16h load_afrag(const half_t* p) {
  v8h lo = *(const v8h*)p;
  v8h hi = *(const v8h*)(p + 8);
  return combine16(lo, hi);
}

// B-matrix fragment (32x16 f16). p = col-row base + chunk base + (lane>>4)*16.
// halves 0..15 <-> k = (lane>>4)*16 + h (contiguous 32 bytes)
__device__ __forceinline__ v16h load_bfrag(const half_t* p) {
  v8h lo = *(const v8h*)p;
  v8h hi = *(const v8h*)(p + 8);
  return combine16(lo, hi);
}

__device__ __forceinline__ v8f wmma_f(v16h a, v16h b, v8f c) {
  return __builtin_amdgcn_wmma_f32_16x16x32_f16(false, a, false, b,
                                                (short)0, c, false, false);
}

// ---------------------------------------------------------------------------
// Projection:  Y[n,e] = sum_d X[n,d]*W[e,d] + b[e]   (f32 in, f16 out)
// grid.x = (rows/64)*8 ; 256 threads = 8 waves ; 64x64 output tile per block
// ---------------------------------------------------------------------------
__global__ __launch_bounds__(256)
void proj_kernel(const float* __restrict__ X, const float* __restrict__ W,
                 const float* __restrict__ bias, half_t* __restrict__ Y) {
  __shared__ __align__(16) half_t xs[64 * 72];
  __shared__ __align__(16) half_t wsld[64 * 72];

  const int tid  = threadIdx.x;
  const int lane = tid & 31;
  const int w    = tid >> 5;
  const int rb   = blockIdx.x >> 3;
  const int cb   = blockIdx.x & 7;
  const int mi   = w >> 1;          // 0..3 row-tile
  const int wc   = w & 1;           // 0..1 col pair
  const int kh   = lane >> 4;
  const int ln   = lane & 15;

  const int li = tid >> 2;          // staging row 0..63
  const int lj = (tid & 3) * 16;    // staging col base

  v8f acc[2] = {zero8(), zero8()};

  for (int kc = 0; kc < 8; ++kc) {  // 512 / 64 K-chunks
    const float* xp = X + (size_t)(rb * 64 + li) * Dm + kc * 64 + lj;
    const float* wp = W + (size_t)(cb * 64 + li) * Dm + kc * 64 + lj;
    half_t* xd = xs   + li * 72 + lj;
    half_t* wd = wsld + li * 72 + lj;
#pragma unroll
    for (int u = 0; u < 16; ++u) xd[u] = (half_t)xp[u];
#pragma unroll
    for (int u = 0; u < 16; ++u) wd[u] = (half_t)wp[u];
    __syncthreads();

#pragma unroll
    for (int sub = 0; sub < 2; ++sub) {     // 64 = 2 x k32
      v16h a = load_afrag(xs + (mi * 16 + ln) * 72 + sub * 32 + kh * 16);
#pragma unroll
      for (int s = 0; s < 2; ++s) {
        v16h b = load_bfrag(wsld + ((wc * 2 + s) * 16 + ln) * 72 + sub * 32 + kh * 16);
        acc[s] = wmma_f(a, b, acc[s]);
      }
    }
    __syncthreads();
  }

#pragma unroll
  for (int s = 0; s < 2; ++s) {
    const int e  = cb * 64 + (wc * 2 + s) * 16 + ln;
    const float bv = bias[e];
#pragma unroll
    for (int r = 0; r < 8; ++r) {
      const int row = rb * 64 + mi * 16 + kh * 8 + r;
      Y[(size_t)row * Dm + e] = (half_t)(acc[s][r] + bv);
    }
  }
}

// ---------------------------------------------------------------------------
// Flash attention. grid.x = B * NQ/64 = 512 ; 256 threads = 8 waves.
// Wave w: query row-tile mi = w>>1 (16 rows), col-half wc = w&1.
//   S: 2 tiles/wave (ni = wc*2+{0,1}); O: 16 tiles/wave (e = h*256+nt*32+wc*16)
// K tile streamed to LDS via async DMA (double-buffered halves); V staged
// transposed in LDS halves so O-GEMM B-fragments are contiguous.
// Dynamic LDS: 128 KB (gfx1250 allows up to 320 KB per workgroup).
// ---------------------------------------------------------------------------
__global__ __launch_bounds__(256, 1)
void attn_kernel(const half_t* __restrict__ Qh, const half_t* __restrict__ Kh,
                 const half_t* __restrict__ Vh, float* __restrict__ Out) {
  extern __shared__ __align__(16) char smem[];
  // byte offsets within the (single, dynamic) LDS block -> LDS_BASE-relative
  constexpr unsigned KS0_OFF = 0;                      // [64 j][264] half d=0..255
  constexpr unsigned KS1_OFF = 64 * 264 * 2;           // 33,792
  constexpr unsigned VTS_OFF = KS1_OFF + 64 * 264 * 2; // 67,584
  constexpr unsigned PB_OFF  = VTS_OFF + 256 * 72 * 2; // 104,448
  constexpr unsigned SB_OFF  = PB_OFF + 64 * 72 * 2;   // 113,664
  constexpr unsigned ST_OFF  = SB_OFF + 64 * 65 * 4;   // 130,304

  half_t* ks0   = (half_t*)(smem + KS0_OFF);
  half_t* ks1   = (half_t*)(smem + KS1_OFF);
  half_t* vts   = (half_t*)(smem + VTS_OFF);
  half_t* pbuf  = (half_t*)(smem + PB_OFF);
  float*  sbuf  = (float*)(smem + SB_OFF);
  float*  rmax_s   = (float*)(smem + ST_OFF);
  float*  rsum_s   = rmax_s + 64;
  float*  rscale_s = rsum_s + 64;

  const int tid  = threadIdx.x;
  const int lane = tid & 31;
  const int w    = tid >> 5;
  const int mi   = w >> 1;
  const int wc   = w & 1;
  const int kh   = lane >> 4;
  const int ln   = lane & 15;

  const int batch = blockIdx.x >> 6;     // NQ/64 = 64 query blocks per batch
  const int qb    = blockIdx.x & 63;
  const float scale = 0.044194173824159216f;   // 1/sqrt(512)

  const size_t qbase  = ((size_t)batch * NQ + qb * 64) * Dm;
  const size_t kvbase = (size_t)batch * NK * Dm;

  // Q A-fragments for this wave's 16 rows, all of D (kept in VGPRs)
  v16h qf[16];
#pragma unroll
  for (int kc = 0; kc < 16; ++kc)
    qf[kc] = load_afrag(Qh + qbase + (size_t)(mi * 16 + ln) * Dm + kc * 32 + kh * 16);

  v8f o[16];
#pragma unroll
  for (int i = 0; i < 16; ++i) o[i] = zero8();

  if (tid < 64) { rmax_s[tid] = -1e30f; rsum_s[tid] = 0.0f; }
  __syncthreads();

  const int vj    = tid >> 2;          // staging: row in KV tile (0..63)
  const int dbase = (tid & 3) * 64;    // staging: 64-wide d segment

  for (int t = 0; t < 64; ++t) {       // NK / 64 KV tiles
    // ---- issue async DMA of K tile t into LDS (both halves, 8+8 x b128) ---
    {
      const half_t* kp = Kh + kvbase + (size_t)(t * 64 + vj) * Dm + dbase;
      const unsigned l0 = KS0_OFF + (unsigned)(vj * 264 + dbase) * 2;
      const unsigned l1 = KS1_OFF + (unsigned)(vj * 264 + dbase) * 2;
#pragma unroll
      for (int u = 0; u < 8; ++u) async_copy16(kp + u * 8, l0 + u * 16);
#pragma unroll
      for (int u = 0; u < 8; ++u) async_copy16(kp + 256 + u * 8, l1 + u * 16);
    }
    if (t + 1 < 64)
      __builtin_prefetch(Vh + kvbase + (size_t)((t + 1) * 64 + vj) * Dm + (tid & 3) * 128, 0, 0);

    // ---- stage V^T half 0 (d = 0..255) while K DMA is in flight ----
    {
      const half_t* vp = Vh + kvbase + (size_t)(t * 64 + vj) * Dm + dbase;
#pragma unroll
      for (int u = 0; u < 8; ++u) {
        v8h d = *(const v8h*)(vp + u * 8);
#pragma unroll
        for (int q = 0; q < 8; ++q) vts[(dbase + u * 8 + q) * 72 + vj] = d[q];
      }
    }

    WAIT_ASYNC(8);       // our 8 ks0 transfers have landed
    __syncthreads();     // all waves' ks0 + vts half0 visible

    // ---- S = Q K^T, d-half 0, from LDS ----
    v8f sc[2] = {zero8(), zero8()};
#pragma unroll
    for (int kc = 0; kc < 8; ++kc) {
#pragma unroll
      for (int s = 0; s < 2; ++s) {
        v16h b = load_bfrag(ks0 + ((wc * 2 + s) * 16 + ln) * 264 + kc * 32 + kh * 16);
        sc[s] = wmma_f(qf[kc], b, sc[s]);
      }
    }

    WAIT_ASYNC(0);       // ks1 transfers landed
    __syncthreads();

    // ---- S = Q K^T, d-half 1, from LDS ----
#pragma unroll
    for (int kc = 0; kc < 8; ++kc) {
#pragma unroll
      for (int s = 0; s < 2; ++s) {
        v16h b = load_bfrag(ks1 + ((wc * 2 + s) * 16 + ln) * 264 + kc * 32 + kh * 16);
        sc[s] = wmma_f(qf[8 + kc], b, sc[s]);
      }
    }
#pragma unroll
    for (int s = 0; s < 2; ++s) {
      const int ni = wc * 2 + s;
#pragma unroll
      for (int r = 0; r < 8; ++r)
        sbuf[(mi * 16 + kh * 8 + r) * 65 + ni * 16 + ln] = sc[s][r] * scale;
    }
    __syncthreads();

    // ---- online softmax (one thread per query row) ----
    if (tid < 64) {
      const int row = tid;
      float mold = rmax_s[row];
      float mmax = mold;
      for (int j = 0; j < 64; ++j) mmax = fmaxf(mmax, sbuf[row * 65 + j]);
      const float c = __expf(mold - mmax);
      float sum = 0.0f;
      for (int j = 0; j < 64; ++j) {
        const float e = __expf(sbuf[row * 65 + j] - mmax);
        sum += e;
        pbuf[row * 72 + j] = (half_t)e;
      }
      rsum_s[row]   = rsum_s[row] * c + sum;
      rmax_s[row]   = mmax;
      rscale_s[row] = c;
    }
    __syncthreads();

    // ---- rescale O, load P fragments ----
    float cr[8];
#pragma unroll
    for (int r = 0; r < 8; ++r) cr[r] = rscale_s[mi * 16 + kh * 8 + r];
#pragma unroll
    for (int nt = 0; nt < 16; ++nt)
#pragma unroll
      for (int r = 0; r < 8; ++r) o[nt][r] *= cr[r];

    v16h pf[2];
#pragma unroll
    for (int jc = 0; jc < 2; ++jc)
      pf[jc] = load_afrag(pbuf + (mi * 16 + ln) * 72 + jc * 32 + kh * 16);

    // ---- O += P V, half 0 (e in [0,256)) ----
#pragma unroll
    for (int nt = 0; nt < 8; ++nt) {
      const int el = nt * 32 + wc * 16;
#pragma unroll
      for (int jc = 0; jc < 2; ++jc) {
        v16h b = load_bfrag(vts + (el + ln) * 72 + jc * 32 + kh * 16);
        o[nt] = wmma_f(pf[jc], b, o[nt]);
      }
    }
    __syncthreads();

    // ---- stage V^T half 1 (d = 256..511) ----
    {
      const half_t* vp = Vh + kvbase + (size_t)(t * 64 + vj) * Dm + 256 + dbase;
#pragma unroll
      for (int u = 0; u < 8; ++u) {
        v8h d = *(const v8h*)(vp + u * 8);
#pragma unroll
        for (int q = 0; q < 8; ++q) vts[(dbase + u * 8 + q) * 72 + vj] = d[q];
      }
    }
    __syncthreads();

    // ---- O += P V, half 1 (e in [256,512)) ----
#pragma unroll
    for (int nt = 0; nt < 8; ++nt) {
      const int el = nt * 32 + wc * 16;
#pragma unroll
      for (int jc = 0; jc < 2; ++jc) {
        v16h b = load_bfrag(vts + (el + ln) * 72 + jc * 32 + kh * 16);
        o[8 + nt] = wmma_f(pf[jc], b, o[8 + nt]);
      }
    }
    __syncthreads();   // protects ks/vts/sbuf/pbuf for the next iteration
  }

  // ---- normalize and write f32 output ----
  float invl[8];
#pragma unroll
  for (int r = 0; r < 8; ++r) invl[r] = 1.0f / rsum_s[mi * 16 + kh * 8 + r];
#pragma unroll
  for (int h = 0; h < 2; ++h) {
#pragma unroll
    for (int nt = 0; nt < 8; ++nt) {
      const int e = h * 256 + nt * 32 + wc * 16 + ln;
#pragma unroll
      for (int r = 0; r < 8; ++r) {
        const size_t row = (size_t)batch * NQ + qb * 64 + mi * 16 + kh * 8 + r;
        Out[row * Dm + e] = o[h * 8 + nt][r] * invl[r];
      }
    }
  }
}

// ---------------------------------------------------------------------------
extern "C" void kernel_launch(void* const* d_in, const int* in_sizes, int n_in,
                              void* d_out, int out_size, void* d_ws, size_t ws_size,
                              hipStream_t stream) {
  const float* x   = (const float*)d_in[0];
  const float* ctx = (const float*)d_in[1];
  const float* Wq  = (const float*)d_in[2];
  const float* bq  = (const float*)d_in[3];
  const float* Wk  = (const float*)d_in[4];
  const float* bk  = (const float*)d_in[5];
  const float* Wv  = (const float*)d_in[6];
  const float* bv  = (const float*)d_in[7];

  const size_t elems = (size_t)Bsz * NQ * Dm;   // 16,777,216 per tensor
  half_t* Qh = (half_t*)d_ws;
  half_t* Kh = Qh + elems;
  half_t* Vh = Kh + elems;

  const int proj_blocks = (Bsz * NQ / 64) * (Dm / 64);   // 4096
  proj_kernel<<<proj_blocks, 256, 0, stream>>>(x,   Wq, bq, Qh);
  proj_kernel<<<proj_blocks, 256, 0, stream>>>(ctx, Wk, bk, Kh);
  proj_kernel<<<proj_blocks, 256, 0, stream>>>(ctx, Wv, bv, Vh);

  const int attn_lds = 131072;   // 128 KB dynamic LDS (gfx1250 max is 320 KB)
  (void)hipFuncSetAttribute((const void*)attn_kernel,
                            hipFuncAttributeMaxDynamicSharedMemorySize, attn_lds);
  attn_kernel<<<Bsz * (NQ / 64), 256, attn_lds, stream>>>(Qh, Kh, Vh, (float*)d_out);
}
